// InterEncoder_22153441313246
// MI455X (gfx1250) — compile-verified, matrix operand order checked by reading, not run on hardware
//
#include <hip/hip_runtime.h>

typedef __attribute__((ext_vector_type(16))) _Float16 v16h;
typedef __attribute__((ext_vector_type(8)))  _Float16 v8h;
typedef __attribute__((ext_vector_type(8)))  float    v8f;

constexpr int H       = 64;     // model dim
constexpr int HD      = 16;     // head dim
constexpr int SPW     = 16;     // samples per wave
constexpr int ROWS    = SPW * 3;  // 48 token-rows per wave (3 full M-tiles)
constexpr int WAVES   = 4;
constexpr int TPB     = WAVES * 32;
constexpr int XPAD    = H + 4;    // f32 row stride (bank-conflict pad, 16B aligned)
constexpr int QKVP    = 200;      // f16 qkv row stride (192 cols + pad)
constexpr int OPAD    = 72;       // f16 attn-out row stride
constexpr int NFRAG_L = 40;       // B fragments per layer: 24 qkv + 8 wo + 8 ffn
constexpr int NFRAG   = 80;
constexpr float LN_EPS = 1e-5f;

struct LayerP {
  const float *w_qkv, *b_qkv, *w_o, *b_o, *ffn_w, *ffn_b;
  const float *n1_g, *n1_b, *n2_g, *n2_b;
};
struct P {
  const float *h_poi, *h_census, *h_traffic;
  const float *mfr_poi, *mfr_census, *mfr_traffic;
  const float *rel_w, *rel_b;
  LayerP L[2];
  float* out;
  int nGroups;
};

__device__ __forceinline__ v8f wmma16(v16h a, v16h b, v8f c) {
  // v_wmma_f32_16x16x32_f16: D = A(16x32 f16) * B(32x16 f16) + C(16x16 f32)
  return __builtin_amdgcn_wmma_f32_16x16x32_f16(false, a, false, b, (short)0, c,
                                                false, false);
}

// A-matrix 16x32 f16 fragment (ISA 7.12.2): lane<16 holds K 0..7 & 16..23 of
// row M=lane; lane>=16 holds K 8..15 & 24..31 of row M=lane-16. kb encodes the
// half-wave K offset (0 or 8) plus kt*32.
__device__ __forceinline__ v16h a_frag_f16(const _Float16* row, int kb) {
  v8h lo = *(const v8h*)(row + kb);
  v8h hi = *(const v8h*)(row + kb + 16);
  return __builtin_shufflevector(lo, hi, 0,1,2,3,4,5,6,7,8,9,10,11,12,13,14,15);
}
__device__ __forceinline__ v16h a_frag_f32(const float* row, int kb) {
  v16h a;
#pragma unroll
  for (int j = 0; j < 8; ++j) {
    a[j]     = (_Float16)row[kb + j];
    a[j + 8] = (_Float16)row[kb + 16 + j];
  }
  return a;
}

__device__ __forceinline__ void ln_rows(const float* tw, float* xw,
                                        const float* gg, const float* bb,
                                        int lane) {
  for (int r = lane; r < ROWS; r += 32) {
    const float* tr = tw + r * XPAD;
    float mu = 0.f;
    for (int k = 0; k < H; ++k) mu += tr[k];
    mu *= (1.0f / H);
    float var = 0.f;
    for (int k = 0; k < H; ++k) { float d = tr[k] - mu; var += d * d; }
    var *= (1.0f / H);
    const float rs = rsqrtf(var + LN_EPS);
    float* xr = xw + r * XPAD;
    for (int k = 0; k < H; ++k) xr[k] = (tr[k] - mu) * rs * gg[k] + bb[k];
  }
}

__global__ __launch_bounds__(TPB) void inter_encoder_kernel(P p) {
  // Weight B-fragments, pre-swizzled: one contiguous 32B read per lane.
  __shared__ __align__(32) _Float16 s_frag[NFRAG][32][16];     // 80 KB
  __shared__ __align__(16) float    s_x[WAVES][ROWS][XPAD];    // 51 KB
  __shared__ __align__(16) float    s_tmp[WAVES][ROWS][XPAD];  // 51 KB
  __shared__ __align__(16) _Float16 s_qkv[WAVES][ROWS][QKVP];  // 75 KB
  __shared__ __align__(16) _Float16 s_o[WAVES][ROWS][OPAD];    // 27 KB
  __shared__ float s_rel[WAVES][ROWS];

  const int tid  = threadIdx.x;
  const int wave = tid >> 5;
  const int lane = tid & 31;

  // ---- Stage weights into WMMA B-fragment layout (once per block) ----
  // B 32x16 f16 layout (mirrors documented 16-bit C layout): lane<16 = col N,
  // holds K pairs {0,1..6,7} then {16..23}; lane>=16 = same cols, K+8 offset.
  for (int idx = tid; idx < NFRAG * 512; idx += TPB) {
    const int f  = idx >> 9;
    const int ln = (idx >> 4) & 31;
    const int e  = idx & 15;
    const int l  = f / NFRAG_L;
    const int fl = f - l * NFRAG_L;
    const float* W; int nt;
    if (fl < 24)      { W = p.L[l].w_qkv; nt = fl >> 1; }
    else if (fl < 32) { W = p.L[l].w_o;   nt = (fl - 24) >> 1; }
    else              { W = p.L[l].ffn_w; nt = (fl - 32) >> 1; }
    const int kt = fl & 1;
    const int v = e >> 1, r = e & 1;
    const int kk  = kt * 32 + ((ln & 16) ? 8 : 0) + (v & 3) * 2 + (v >> 2) * 16 + r;
    const int col = nt * 16 + (ln & 15);
    // W stored (out,in) row-major; B = W^T so B[k][n] = W[n*H + k]
    s_frag[f][ln][e] = (_Float16)W[col * H + kk];
  }
  __syncthreads();

  float*    xw = &s_x[wave][0][0];
  float*    tw = &s_tmp[wave][0][0];
  _Float16* qw = &s_qkv[wave][0][0];
  _Float16* ow = &s_o[wave][0][0];
  float*    rw = &s_rel[wave][0];

  const int laneN  = lane & 15;
  const int hiHalf = (lane & 16) ? 8 : 0;
  const int gstep  = gridDim.x * WAVES;

  for (int g = blockIdx.x * WAVES + wave; g < p.nGroups; g += gstep) {
    const long sbase = (long)g * SPW;

    // Prefetch next chunk (global_prefetch_b8), 128B/lane covers 16x64 f32.
    const long sN = sbase + (long)gstep * SPW;
    if (sN < (long)p.nGroups * SPW) {
      __builtin_prefetch(p.h_poi     + sN * H + lane * 32, 0, 1);
      __builtin_prefetch(p.h_census  + sN * H + lane * 32, 0, 1);
      __builtin_prefetch(p.h_traffic + sN * H + lane * 32, 0, 1);
    }

    // ---- Phase 0: load x rows (f32) and per-token relevance gate ----
    for (int i = lane; i < ROWS * 16; i += 32) {
      const int row = i >> 4, q4 = i & 15;
      const long s = sbase + row / 3;
      const int  m = row % 3;
      const float* sp = (m == 0) ? p.h_poi : (m == 1) ? p.h_census : p.h_traffic;
      const float4 val = ((const float4*)(sp + s * H))[q4];
      ((float4*)(xw + row * XPAD))[q4] = val;
    }
    for (int i = lane; i < ROWS; i += 32) {
      const long s = sbase + i / 3;
      const int  m = i % 3;
      const float* mp = (m == 0) ? p.mfr_poi : (m == 1) ? p.mfr_census : p.mfr_traffic;
      const float z = mp[s * 2] * p.rel_w[0] + mp[s * 2 + 1] * p.rel_w[1] + p.rel_b[0];
      rw[i] = 1.0f / (1.0f + expf(-z));
    }

    for (int l = 0; l < 2; ++l) {
      const LayerP& Lw = p.L[l];
      const int fb = l * NFRAG_L;

      // ---- QKV projection: (48x64) @ Wqkv^T -> s_qkv f16 (48x192) ----
      for (int mt = 0; mt < 3; ++mt) {
        const float* xrow = xw + (mt * 16 + laneN) * XPAD;
        const v16h a0 = a_frag_f32(xrow, hiHalf);
        const v16h a1 = a_frag_f32(xrow, 32 + hiHalf);
        for (int nt = 0; nt < 12; ++nt) {
          v8f acc = {};
          const v16h b0 = *(const v16h*)&s_frag[fb + nt * 2 + 0][lane][0];
          const v16h b1 = *(const v16h*)&s_frag[fb + nt * 2 + 1][lane][0];
          acc = wmma16(a0, b0, acc);
          acc = wmma16(a1, b1, acc);
          const int col   = nt * 16 + laneN;
          const float bias = Lw.b_qkv[col];
          _Float16* dst = qw + (mt * 16 + hiHalf) * QKVP + col;
#pragma unroll
          for (int v = 0; v < 8; ++v)
            dst[v * QKVP] = (_Float16)(acc[v] + bias);
        }
      }

      // ---- Attention: 16 samples x 4 heads x 3 queries = 192 tasks ----
#pragma unroll
      for (int t6 = 0; t6 < 6; ++t6) {
        const int task = lane + 32 * t6;
        const int s  = task / 12;
        const int rm = task - s * 12;
        const int hh = rm / 3;
        const int qi = rm - hh * 3;
        const int r0 = s * 3;
        const _Float16* qp = qw + (r0 + qi) * QKVP + hh * HD;
        float qv[HD];
#pragma unroll
        for (int j = 0; j < HD; ++j) qv[j] = (float)qp[j];
        float sc[3];
#pragma unroll
        for (int t = 0; t < 3; ++t) {
          const _Float16* kp = qw + (r0 + t) * QKVP + H + hh * HD;
          float d = 0.f;
#pragma unroll
          for (int j = 0; j < HD; ++j) d += qv[j] * (float)kp[j];
          sc[t] = d * 0.25f;  // 1/sqrt(16)
        }
        const float mx = fmaxf(sc[0], fmaxf(sc[1], sc[2]));
        const float e0 = expf(sc[0] - mx), e1 = expf(sc[1] - mx), e2 = expf(sc[2] - mx);
        const float inv = 1.0f / (e0 + e1 + e2);
        const float a0w = e0 * inv, a1w = e1 * inv, a2w = e2 * inv;
        const _Float16* v0 = qw + (r0 + 0) * QKVP + 2 * H + hh * HD;
        const _Float16* v1 = v0 + QKVP;
        const _Float16* v2 = v1 + QKVP;
        _Float16* op = ow + (r0 + qi) * OPAD + hh * HD;
#pragma unroll
        for (int j = 0; j < HD; ++j)
          op[j] = (_Float16)(a0w * (float)v0[j] + a1w * (float)v1[j] +
                             a2w * (float)v2[j]);
      }

      // ---- Out-proj + rel gate + residual -> s_tmp (f32) ----
      for (int mt = 0; mt < 3; ++mt) {
        const _Float16* orow = ow + (mt * 16 + laneN) * OPAD;
        const v16h a0 = a_frag_f16(orow, hiHalf);
        const v16h a1 = a_frag_f16(orow, 32 + hiHalf);
        for (int nt = 0; nt < 4; ++nt) {
          v8f acc = {};
          const v16h b0 = *(const v16h*)&s_frag[fb + 24 + nt * 2 + 0][lane][0];
          const v16h b1 = *(const v16h*)&s_frag[fb + 24 + nt * 2 + 1][lane][0];
          acc = wmma16(a0, b0, acc);
          acc = wmma16(a1, b1, acc);
          const int col   = nt * 16 + laneN;
          const float bias = Lw.b_o[col];
          const int rbase = mt * 16 + hiHalf;
#pragma unroll
          for (int v = 0; v < 8; ++v) {
            const int row = rbase + v;
            tw[row * XPAD + col] =
                (acc[v] + bias) * rw[row] + xw[row * XPAD + col];
          }
        }
      }

      // ---- LayerNorm 1: s_tmp -> s_x ----
      ln_rows(tw, xw, Lw.n1_g, Lw.n1_b, lane);

      // ---- FFN: relu(x @ Wf^T + bf) + residual -> s_tmp ----
      for (int mt = 0; mt < 3; ++mt) {
        const float* xrow = xw + (mt * 16 + laneN) * XPAD;
        const v16h a0 = a_frag_f32(xrow, hiHalf);
        const v16h a1 = a_frag_f32(xrow, 32 + hiHalf);
        for (int nt = 0; nt < 4; ++nt) {
          v8f acc = {};
          const v16h b0 = *(const v16h*)&s_frag[fb + 32 + nt * 2 + 0][lane][0];
          const v16h b1 = *(const v16h*)&s_frag[fb + 32 + nt * 2 + 1][lane][0];
          acc = wmma16(a0, b0, acc);
          acc = wmma16(a1, b1, acc);
          const int col   = nt * 16 + laneN;
          const float bias = Lw.ffn_b[col];
          const int rbase = mt * 16 + hiHalf;
#pragma unroll
          for (int v = 0; v < 8; ++v) {
            const int row = rbase + v;
            tw[row * XPAD + col] =
                fmaxf(acc[v] + bias, 0.0f) + xw[row * XPAD + col];
          }
        }
      }

      // ---- LayerNorm 2: s_tmp -> s_x ----
      ln_rows(tw, xw, Lw.n2_g, Lw.n2_b, lane);
    }

    // ---- Store: rows are contiguous in (B,3,H) at sbase*3*H ----
    for (int i = lane; i < ROWS * 16; i += 32) {
      const int row = i >> 4, q4 = i & 15;
      const float4 val = ((const float4*)(xw + row * XPAD))[q4];
      ((float4*)(p.out + (sbase * 3 + row) * H))[q4] = val;
    }
  }
}

extern "C" void kernel_launch(void* const* d_in, const int* in_sizes, int n_in,
                              void* d_out, int out_size, void* d_ws, size_t ws_size,
                              hipStream_t stream) {
  (void)n_in; (void)out_size; (void)d_ws; (void)ws_size;
  P p;
  p.h_poi       = (const float*)d_in[0];
  p.h_census    = (const float*)d_in[1];
  p.h_traffic   = (const float*)d_in[2];
  p.mfr_poi     = (const float*)d_in[3];
  p.mfr_census  = (const float*)d_in[4];
  p.mfr_traffic = (const float*)d_in[5];
  p.rel_w       = (const float*)d_in[6];
  p.rel_b       = (const float*)d_in[7];
  int idx = 8;
  for (int l = 0; l < 2; ++l) {
    p.L[l].w_qkv = (const float*)d_in[idx++];
    p.L[l].b_qkv = (const float*)d_in[idx++];
    p.L[l].w_o   = (const float*)d_in[idx++];
    p.L[l].b_o   = (const float*)d_in[idx++];
    p.L[l].ffn_w = (const float*)d_in[idx++];
    p.L[l].ffn_b = (const float*)d_in[idx++];
    p.L[l].n1_g  = (const float*)d_in[idx++];
    p.L[l].n1_b  = (const float*)d_in[idx++];
    p.L[l].n2_g  = (const float*)d_in[idx++];
    p.L[l].n2_b  = (const float*)d_in[idx++];
  }
  p.out = (float*)d_out;
  const int B = in_sizes[0] / H;
  p.nGroups = B / SPW;
  int grid = (p.nGroups + WAVES - 1) / WAVES;
  if (grid > 4096) grid = 4096;
  if (grid < 1) grid = 1;
  hipLaunchKernelGGL(inter_encoder_kernel, dim3(grid), dim3(TPB), 0, stream, p);
}
